// MultiHeadAttention_81681688035914
// MI455X (gfx1250) — matrix operand order repaired and checked by reading, not compiled, not run on hardware
//
#include <hip/hip_runtime.h>

// ---------------------------------------------------------------------------
// Causal multi-head attention forward for gfx1250 (MI455X), bf16 WMMA path.
//   B=4, T=2048, C=1024, H=16, d=64
// Stages:
//   1) cast/transpose weights+activations to bf16 (w stored [N,K] so B-frags
//      are K-contiguous)
//   2) QKV GEMM: 32x64 C-tile per wave (2 A-frags x 4 B-frags, 8 WMMA/K-step);
//      scatter q(scaled)/k -> [BH,T,d], v -> transposed [BH,d,T]
//   3) flash-attention: per-wave 32-row query tile, 32-wide key blocks,
//      online softmax, P relayout through LDS (s_wait_dscnt), O accum in f32
//   4) out-proj GEMM (same 32x64 tiling) + bias, f32 output
// ---------------------------------------------------------------------------

#define BATCH   4
#define SEQ     2048
#define DM      1024            // d_model
#define NH      16              // heads
#define HD      64              // head dim
#define MROWS   (BATCH*SEQ)     // 8192
#define NQKV    (3*DM)          // 3072

typedef __attribute__((ext_vector_type(16))) __bf16   v16bf;
typedef __attribute__((ext_vector_type(8)))  float    v8f;
typedef __attribute__((ext_vector_type(8)))  unsigned v8u;
typedef unsigned short u16;     // raw bf16 storage

__device__ __forceinline__ u16 f2bf(float f) {
  unsigned u = __builtin_bit_cast(unsigned, f);
  u += 0x7FFFu + ((u >> 16) & 1u);          // round-to-nearest-even
  return (u16)(u >> 16);
}

__device__ __forceinline__ v8f wmma_bf16(v16bf a, v16bf b, v8f c) {
  // (neg_a, A, neg_b, B, c_mod, C, reuse_a, reuse_b)
  return __builtin_amdgcn_wmma_f32_16x16x32_bf16(false, a, false, b,
                                                 (short)0, c, false, false);
}

// A fragment: 16(M) x 32(K) bf16 tile, row-major, leading dim `ld` elements.
// Layout (16-bit A): lanes 0-15 row M=lane, K=0..15 sequential in v0..7;
// lanes 16-31 same rows with K=16..31 (matches load_b).
__device__ __forceinline__ v16bf load_a(const u16* p, int ld) {
  const int lane = threadIdx.x & 31;
  const int half = lane >> 4;
  const int r    = lane & 15;
  const u16* row = p + r * ld + half * 16;
  v8u raw;
#pragma unroll
  for (int i = 0; i < 8; ++i) {
    const int kk = 2 * i;
    raw[i] = *(const unsigned*)(row + kk);
  }
  return __builtin_bit_cast(v16bf, raw);
}

// B fragment: 32(K) x 16(N) bf16. Source is B^T stored row-major (column n of
// B is a contiguous row of 32 K-values); ld = K-stride of B^T.
// Layout used here (16-bit B, same k order as load_a): lanes 0-15 col N=lane with K=0..15 sequential in
// v0..7 (2 per VGPR); lanes 16-31 hold K=16..31.
__device__ __forceinline__ v16bf load_b(const u16* p, int ld) {
  const int lane = threadIdx.x & 31;
  const int half = lane >> 4;
  const int r    = lane & 15;
  const u16* row = p + r * ld + half * 16;   // 16 contiguous bf16 = 32 bytes
  const uint4* q = (const uint4*)row;
  uint4 x = q[0];
  uint4 y = q[1];
  v8u raw;
  raw[0] = x.x; raw[1] = x.y; raw[2] = x.z; raw[3] = x.w;
  raw[4] = y.x; raw[5] = y.y; raw[6] = y.z; raw[7] = y.w;
  return __builtin_bit_cast(v16bf, raw);
}

// ---------------------------------------------------------------------------
// Stage 1: casts
// ---------------------------------------------------------------------------
__global__ __launch_bounds__(256) void cast_f32_bf16(const float* __restrict__ x,
                                                     u16* __restrict__ y, int n) {
  int i = blockIdx.x * 256 + threadIdx.x;
  if (i < n) y[i] = f2bf(x[i]);
}

// w[K][N] (f32) -> wT[N][K] (bf16)
__global__ __launch_bounds__(256) void transpose_cast(const float* __restrict__ w,
                                                      u16* __restrict__ wT,
                                                      int K, int N) {
  int i = blockIdx.x * 256 + threadIdx.x;
  if (i < K * N) {
    int nn = i / K;
    int kk = i - nn * K;
    wT[i] = f2bf(w[kk * N + nn]);
  }
}

// ---------------------------------------------------------------------------
// Stage 2: QKV GEMM.  qkv[m, ng] = x[m, :] . w_qkv[:, ng]
// grid = (NQKV/64, MROWS/256), block = 256 (8 waves, one 32x64 tile each)
// Per K-step: 2 A-frags + 4 B-frags -> 8 WMMAs (register-level reuse).
// ---------------------------------------------------------------------------
__global__ __launch_bounds__(256) void qkv_gemm(const u16* __restrict__ xb,
                                                const u16* __restrict__ wqkvT,
                                                u16* __restrict__ qb,
                                                u16* __restrict__ kb,
                                                u16* __restrict__ vT) {
  const int wave = threadIdx.x >> 5;
  const int lane = threadIdx.x & 31;
  const int half = lane >> 4;
  const int nlo  = lane & 15;

  const int m0 = (blockIdx.y * 8 + wave) * 32;   // 32 output rows
  const int n0 = blockIdx.x * 64;                // 64 output cols

  v8f acc[2][4] = {};
  for (int k0 = 0; k0 < DM; k0 += 32) {
    // WGP-scope prefetch (locality 3): pull next K-tiles into near caches,
    // not just GL2 (SYS-scope prefetch does no WGP-level prefetch).
    __builtin_prefetch(xb + (size_t)(m0 + nlo) * DM + k0 + 256, 0, 3);
    __builtin_prefetch(wqkvT + (size_t)(n0 + nlo) * DM + k0 + 256, 0, 3);
    v16bf a0 = load_a(xb + (size_t)(m0)      * DM + k0, DM);
    v16bf a1 = load_a(xb + (size_t)(m0 + 16) * DM + k0, DM);
#pragma unroll
    for (int j = 0; j < 4; ++j) {
      v16bf b = load_b(wqkvT + (size_t)(n0 + 16 * j) * DM + k0, DM);
      acc[0][j] = wmma_bf16(a0, b, acc[0][j]);
      acc[1][j] = wmma_bf16(a1, b, acc[1][j]);
    }
  }

  // scatter: ng -> (section, head, dd)
#pragma unroll
  for (int j = 0; j < 4; ++j) {
    const int ng  = n0 + 16 * j + nlo;
    const int sec = ng >> 10;        // 0=q 1=k 2=v
    const int c   = ng & 1023;
    const int h   = c >> 6;
    const int dd  = c & 63;
#pragma unroll
    for (int mt = 0; mt < 2; ++mt) {
#pragma unroll
      for (int i = 0; i < 8; ++i) {
        const int m  = m0 + mt * 16 + i + 8 * half;  // row in [0, MROWS)
        const int b_ = m >> 11;                      // batch
        const int t  = m & (SEQ - 1);
        const int bh = b_ * NH + h;
        const float v = acc[mt][j][i];
        if (sec == 0) {
          qb[((size_t)bh * SEQ + t) * HD + dd] = f2bf(v * 0.125f); // 1/sqrt(64)
        } else if (sec == 1) {
          kb[((size_t)bh * SEQ + t) * HD + dd] = f2bf(v);
        } else {
          vT[((size_t)bh * HD + dd) * SEQ + t] = f2bf(v);          // transposed
        }
      }
    }
  }
}

// ---------------------------------------------------------------------------
// Stage 3: causal flash attention.
// grid = (SEQ/256, BATCH*NH), block = 256. Each wave: one 32-row query tile
// (2 row-subtiles). Per 32-key block: 8 WMMAs for S, 8 WMMAs for O.
// No cross-wave sync (key-loop trip count is wave-dependent).
// ---------------------------------------------------------------------------
__global__ __launch_bounds__(256) void flash_attn(const u16* __restrict__ qb,
                                                  const u16* __restrict__ kb,
                                                  const u16* __restrict__ vT,
                                                  u16* __restrict__ attnb) {
  __shared__ u16 ldsP[8][32 * 32];   // per-wave P-tile staging (16 KB)

  const int wave = threadIdx.x >> 5;
  const int lane = threadIdx.x & 31;
  const int half = lane >> 4;
  const int nlo  = lane & 15;

  const int bh  = blockIdx.y;
  const int qb0 = blockIdx.x * 256 + wave * 32;

  const u16* qh = qb + (size_t)bh * SEQ * HD;
  const u16* kh = kb + (size_t)bh * SEQ * HD;
  const u16* vh = vT + (size_t)bh * HD * SEQ;

  // query fragments: 32 x 64 = 2 row-subtiles x 2 K-chunks (q pre-scaled)
  v16bf aq00 = load_a(qh + (size_t)(qb0)      * HD + 0,  HD);
  v16bf aq01 = load_a(qh + (size_t)(qb0)      * HD + 32, HD);
  v16bf aq10 = load_a(qh + (size_t)(qb0 + 16) * HD + 0,  HD);
  v16bf aq11 = load_a(qh + (size_t)(qb0 + 16) * HD + 32, HD);

  v8f o[2][4] = {};
  float mx[2][8], ls[2][8];
#pragma unroll
  for (int rt = 0; rt < 2; ++rt)
#pragma unroll
    for (int i = 0; i < 8; ++i) { mx[rt][i] = -1e30f; ls[rt][i] = 0.0f; }

  for (int j0 = 0; j0 < qb0 + 32; j0 += 32) {    // 32-wide key blocks
    // S = q . k^T : 2 row-subtiles x 2 col-subtiles x 2 K-chunks = 8 WMMAs
    v16bf k00 = load_b(kh + (size_t)(j0)      * HD + 0,  HD);
    v16bf k01 = load_b(kh + (size_t)(j0)      * HD + 32, HD);
    v16bf k10 = load_b(kh + (size_t)(j0 + 16) * HD + 0,  HD);
    v16bf k11 = load_b(kh + (size_t)(j0 + 16) * HD + 32, HD);
    v8f s[2][2] = {};
    s[0][0] = wmma_bf16(aq00, k00, s[0][0]);
    s[0][0] = wmma_bf16(aq01, k01, s[0][0]);
    s[0][1] = wmma_bf16(aq00, k10, s[0][1]);
    s[0][1] = wmma_bf16(aq01, k11, s[0][1]);
    s[1][0] = wmma_bf16(aq10, k00, s[1][0]);
    s[1][0] = wmma_bf16(aq11, k01, s[1][0]);
    s[1][1] = wmma_bf16(aq10, k10, s[1][1]);
    s[1][1] = wmma_bf16(aq11, k11, s[1][1]);

    // causal mask + online softmax (per output row; 16-lane reductions stay
    // inside each half-wave for xor masks 1,2,4,8)
#pragma unroll
    for (int rt = 0; rt < 2; ++rt) {
#pragma unroll
      for (int i = 0; i < 8; ++i) {
        const int m  = rt * 16 + i + 8 * half;
        const int qi = qb0 + m;
        const float a = (j0 + nlo      <= qi) ? s[rt][0][i] : -1e30f;
        const float b = (j0 + 16 + nlo <= qi) ? s[rt][1][i] : -1e30f;
        float v = fmaxf(a, b);
#pragma unroll
        for (int msk = 1; msk < 16; msk <<= 1) v = fmaxf(v, __shfl_xor(v, msk, 32));
        const float mnew  = fmaxf(mx[rt][i], v);
        const float scale = __expf(mx[rt][i] - mnew);
        const float p0 = __expf(a - mnew);
        const float p1 = __expf(b - mnew);
        float rs = p0 + p1;
#pragma unroll
        for (int msk = 1; msk < 16; msk <<= 1) rs += __shfl_xor(rs, msk, 32);
        ls[rt][i] = ls[rt][i] * scale + rs;
        mx[rt][i] = mnew;
        o[rt][0][i] *= scale; o[rt][1][i] *= scale;
        o[rt][2][i] *= scale; o[rt][3][i] *= scale;
        // stage P (C-layout) into LDS row-major 32x32 for A-frag reload
        ldsP[wave][m * 32 + nlo]      = f2bf(p0);
        ldsP[wave][m * 32 + 16 + nlo] = f2bf(p1);
      }
    }
    asm volatile("s_wait_dscnt 0" ::: "memory");   // LDS RAW, wave-local

    v16bf pa0 = load_a(&ldsP[wave][0],       32);
    v16bf pa1 = load_a(&ldsP[wave][16 * 32], 32);
    // O += P . V ; V column n lives as contiguous row of vT
#pragma unroll
    for (int ch = 0; ch < 4; ++ch) {
      v16bf vv = load_b(vh + (size_t)(ch * 16) * SEQ + j0, SEQ);
      o[0][ch] = wmma_bf16(pa0, vv, o[0][ch]);
      o[1][ch] = wmma_bf16(pa1, vv, o[1][ch]);
    }
  }

  // normalize and store to attn[B*T, C] (bf16, feeds out-proj GEMM)
  const int b_ = bh >> 4;
  const int h  = bh & 15;
#pragma unroll
  for (int rt = 0; rt < 2; ++rt) {
#pragma unroll
    for (int i = 0; i < 8; ++i) {
      const int m    = rt * 16 + i + 8 * half;
      const float rn = 1.0f / ls[rt][i];
      const size_t row = (size_t)(b_ * SEQ + qb0 + m) * DM + h * HD + nlo;
      attnb[row + 0 * 16] = f2bf(o[rt][0][i] * rn);
      attnb[row + 1 * 16] = f2bf(o[rt][1][i] * rn);
      attnb[row + 2 * 16] = f2bf(o[rt][2][i] * rn);
      attnb[row + 3 * 16] = f2bf(o[rt][3][i] * rn);
    }
  }
}

// ---------------------------------------------------------------------------
// Stage 4: output projection. out[m,n] = attn[m,:] . w_out[:,n] + b_out[n]
// grid = (DM/64, MROWS/256), block = 256; 32x64 tile per wave.
// ---------------------------------------------------------------------------
__global__ __launch_bounds__(256) void out_proj(const u16* __restrict__ attnb,
                                                const u16* __restrict__ woT,
                                                const float* __restrict__ bias,
                                                float* __restrict__ out) {
  const int wave = threadIdx.x >> 5;
  const int lane = threadIdx.x & 31;
  const int half = lane >> 4;
  const int nlo  = lane & 15;

  const int m0 = (blockIdx.y * 8 + wave) * 32;
  const int n0 = blockIdx.x * 64;

  v8f acc[2][4] = {};
  for (int k0 = 0; k0 < DM; k0 += 32) {
    __builtin_prefetch(attnb + (size_t)(m0 + nlo) * DM + k0 + 256, 0, 3);
    __builtin_prefetch(woT + (size_t)(n0 + nlo) * DM + k0 + 256, 0, 3);
    v16bf a0 = load_a(attnb + (size_t)(m0)      * DM + k0, DM);
    v16bf a1 = load_a(attnb + (size_t)(m0 + 16) * DM + k0, DM);
#pragma unroll
    for (int j = 0; j < 4; ++j) {
      v16bf b = load_b(woT + (size_t)(n0 + 16 * j) * DM + k0, DM);
      acc[0][j] = wmma_bf16(a0, b, acc[0][j]);
      acc[1][j] = wmma_bf16(a1, b, acc[1][j]);
    }
  }

#pragma unroll
  for (int j = 0; j < 4; ++j) {
    const int n  = n0 + 16 * j + nlo;
    const float bv = bias[n];
#pragma unroll
    for (int mt = 0; mt < 2; ++mt) {
#pragma unroll
      for (int i = 0; i < 8; ++i) {
        const int m = m0 + mt * 16 + i + 8 * half;
        out[(size_t)m * DM + n] = acc[mt][j][i] + bv;
      }
    }
  }
}

// ---------------------------------------------------------------------------
extern "C" void kernel_launch(void* const* d_in, const int* in_sizes, int n_in,
                              void* d_out, int out_size, void* d_ws, size_t ws_size,
                              hipStream_t stream) {
  const float* x     = (const float*)d_in[0];   // [B,T,C]
  const float* w_qkv = (const float*)d_in[1];   // [C,3C]
  const float* w_out = (const float*)d_in[2];   // [C,C]
  const float* b_out = (const float*)d_in[3];   // [C]
  float* out = (float*)d_out;                   // [B,T,C]

  // workspace partition (bf16 stored as u16); ~88 MB total
  char* w = (char*)d_ws;
  u16* xb    = (u16*)w; w += (size_t)MROWS * DM * 2;        // 16 MB
  u16* wqkvT = (u16*)w; w += (size_t)NQKV * DM * 2;         //  6 MB
  u16* woT   = (u16*)w; w += (size_t)DM * DM * 2;           //  2 MB
  u16* qb    = (u16*)w; w += (size_t)BATCH * NH * SEQ * HD * 2;  // 16 MB
  u16* kb    = (u16*)w; w += (size_t)BATCH * NH * SEQ * HD * 2;  // 16 MB
  u16* vT    = (u16*)w; w += (size_t)BATCH * NH * HD * SEQ * 2;  // 16 MB
  u16* attnb = (u16*)w; w += (size_t)MROWS * DM * 2;             // 16 MB
  (void)ws_size; (void)n_in; (void)in_sizes; (void)out_size;

  // Stage 1: casts / transposes
  {
    int n = MROWS * DM;
    cast_f32_bf16<<<(n + 255) / 256, 256, 0, stream>>>(x, xb, n);
    int t1 = DM * NQKV;
    transpose_cast<<<(t1 + 255) / 256, 256, 0, stream>>>(w_qkv, wqkvT, DM, NQKV);
    int t2 = DM * DM;
    transpose_cast<<<(t2 + 255) / 256, 256, 0, stream>>>(w_out, woT, DM, DM);
  }

  // Stage 2: QKV projection GEMM (M=8192, N=3072, K=1024)
  qkv_gemm<<<dim3(NQKV / 64, MROWS / 256), 256, 0, stream>>>(xb, wqkvT, qb, kb, vT);

  // Stage 3: causal flash attention
  flash_attn<<<dim3(SEQ / 256, BATCH * NH), 256, 0, stream>>>(qb, kb, vT, attnb);

  // Stage 4: output projection + bias (M=8192, N=1024, K=1024)
  out_proj<<<dim3(DM / 64, MROWS / 256), 256, 0, stream>>>(attnb, woT, b_out, out);
}
